// SE3MP_87548613361690
// MI455X (gfx1250) — matrix operand order repaired and checked by reading, not compiled, not run on hardware
//
#include <hip/hip_runtime.h>
#include <hip/hip_bf16.h>

typedef __attribute__((ext_vector_type(16))) _Float16 v16h;
typedef __attribute__((ext_vector_type(8)))  _Float16 v8h;
typedef __attribute__((ext_vector_type(8)))  float    v8f;

#define MUL 8
#define HIDDEN 64
#define WNUM 256
#define ALPHA 0.25f                 // 1/sqrt(2*MUL)
#define INV_SQRT3 0.57735026919f
#define SKIP_NORM 0.35355339059f    // 1/sqrt(MUL)
#define WAVES_PER_BLOCK 2
#define EDGES_PER_BLOCK (WAVES_PER_BLOCK * 16)

// ---------------------------------------------------------------------------
// Pack mlp_w1 (16x64, K padded to 32) and mlp_w2 (64x256) into per-lane WMMA
// B-fragment layout (f16).  Entry layout in ws (halves):
//   [0 .. 128)   : w1 frags  [ntile(4)][lane(32)] x 16 halves
//   [128 .. 1152): w2 frags  [kstep(2)][ntile(16)][lane(32)] x 16 halves
// B element e at lane l:  K = kbase + (e<8 ? (l>=16)*8 + e : 16 + (l>=16)*8 + e-8) ,  N = 16*ntile + (l&15)
// ---------------------------------------------------------------------------
__global__ void se3mp_prep_frags(const float* __restrict__ w1,
                                 const float* __restrict__ w2,
                                 _Float16* __restrict__ wsh) {
  for (int ent = blockIdx.x * blockDim.x + threadIdx.x; ent < 1152;
       ent += gridDim.x * blockDim.x) {
    if (ent < 128) {
      int ntile = ent >> 5, lane = ent & 31;
      int hg = lane >> 4, nl = lane & 15;
      int N = ntile * 16 + nl;
      #pragma unroll
      for (int e = 0; e < 16; ++e) {
        int K = (e < 8) ? (hg * 8 + e) : (16 + hg * 8 + (e - 8));
        float v = (K < 16) ? w1[K * HIDDEN + N] : 0.0f;
        wsh[ent * 16 + e] = (_Float16)v;
      }
    } else {
      int r = ent - 128;                 // 0..1023
      int kstep = r >> 9, ntile = (r >> 5) & 15, lane = r & 31;
      int hg = lane >> 4, nl = lane & 15;
      int N = ntile * 16 + nl;
      #pragma unroll
      for (int e = 0; e < 16; ++e) {
        int K = kstep * 32 + ((e < 8) ? (hg * 8 + e) : (16 + hg * 8 + (e - 8)));
        wsh[ent * 16 + e] = (_Float16)w2[K * WNUM + N];
      }
    }
  }
}

// ---------------------------------------------------------------------------
// Per-node skip connection; initializes d_out (runs before edge atomics).
// ---------------------------------------------------------------------------
__global__ void se3mp_skip(const float* __restrict__ node_feats,
                           const float* __restrict__ skip_w0,
                           const float* __restrict__ skip_w1,
                           float* __restrict__ out, int n_nodes) {
  int n = blockIdx.x * blockDim.x + threadIdx.x;
  if (n >= n_nodes) return;
  const float* x = node_feats + (long long)n * 32;
  float* o = out + (long long)n * 32;
  #pragma unroll
  for (int j = 0; j < MUL; ++j) {
    float a = 0.f;
    #pragma unroll
    for (int i = 0; i < MUL; ++i) a += x[i] * skip_w0[i * MUL + j];
    o[j] = a * SKIP_NORM;
  }
  #pragma unroll
  for (int j = 0; j < MUL; ++j) {
    #pragma unroll
    for (int m = 0; m < 3; ++m) {
      float a = 0.f;
      #pragma unroll
      for (int i = 0; i < MUL; ++i) a += x[8 + i * 3 + m] * skip_w1[i * MUL + j];
      o[8 + j * 3 + m] = a * SKIP_NORM;
    }
  }
}

// ---------------------------------------------------------------------------
// Edge kernel: WMMA edge-MLP (16 edges per wave) + SE(3) contraction + scatter
// ---------------------------------------------------------------------------
__global__ __launch_bounds__(EDGES_PER_BLOCK * 2)
void se3mp_edge(const float* __restrict__ node_feats,
                const float* __restrict__ edge_feat,
                const float* __restrict__ edge_sh,
                const long long* __restrict__ edge_index,
                const float* __restrict__ b1,
                const float* __restrict__ b2,
                const _Float16* __restrict__ wsh,
                float* __restrict__ out,
                int n_edges) {
  __shared__ __align__(16) _Float16 h_lds[WAVES_PER_BLOCK][16][HIDDEN];
  __shared__ __align__(16) float    w_lds[WAVES_PER_BLOCK][16][WNUM];

  const int tid  = threadIdx.x;
  const int wave = tid >> 5;
  const int lane = tid & 31;
  const int hg   = lane >> 4;    // lane half-group
  const int mrow = lane & 15;    // M row / N column-within-tile

  const long long tile_base = (long long)blockIdx.x * EDGES_PER_BLOCK + wave * 16;
  const long long my_edge   = tile_base + mrow;

  // ---- GEMM1 A fragment: edge_feat row (K=16 real, padded to 32) ----------
  v16h a1;
  #pragma unroll
  for (int e = 0; e < 16; ++e) a1[e] = (_Float16)0.0f;
  if (my_edge < n_edges) {
    const float* row = edge_feat + my_edge * 16 + hg * 8;  // 8 contiguous f32
    #pragma unroll
    for (int e = 0; e < 8; ++e) a1[e] = (_Float16)row[e];
  }

  // ---- GEMM1: h = silu(edge_feat @ w1 + b1) -------------------------------
  #pragma unroll
  for (int n = 0; n < 4; ++n) {
    v8f acc = {};
    v16h b = *(const v16h*)(wsh + ((long long)(n * 32 + lane)) * 16);
    acc = __builtin_amdgcn_wmma_f32_16x16x32_f16(false, a1, false, b,
                                                 (short)0, acc, false, false);
    int col = n * 16 + mrow;
    float bb = b1[col];
    #pragma unroll
    for (int r = 0; r < 8; ++r) {
      float x = acc[r] + bb;
      // silu via hardware v_rcp_f32 (avoid IEEE division sequence)
      float s = x * __builtin_amdgcn_rcpf(1.0f + __expf(-x));
      h_lds[wave][r + hg * 8][col] = (_Float16)s;
    }
  }
  __syncthreads();

  // ---- Re-marshal h into A fragments (two K-steps of 32) ------------------
  v16h a2[2];
  #pragma unroll
  for (int k = 0; k < 2; ++k) {
    const _Float16* hr = &h_lds[wave][mrow][k * 32 + hg * 8];
    v8h lo = *(const v8h*)hr;          // K = k*32 + hg*8 .. +7
    v8h hi = *(const v8h*)(hr + 16);   // K = k*32 + 16 + hg*8 .. +7
    #pragma unroll
    for (int e = 0; e < 8; ++e) { a2[k][e] = lo[e]; a2[k][8 + e] = hi[e]; }
  }

  // ---- GEMM2: w = h @ w2 + b2  (16 N-tiles x 2 K-steps) -------------------
  const _Float16* w2f = wsh + 128 * 16;
  #pragma unroll
  for (int n = 0; n < 16; ++n) {
    v8f acc = {};
    #pragma unroll
    for (int k = 0; k < 2; ++k) {
      v16h b = *(const v16h*)(w2f + ((long long)((k * 16 + n) * 32 + lane)) * 16);
      acc = __builtin_amdgcn_wmma_f32_16x16x32_f16(false, a2[k], false, b,
                                                   (short)0, acc, false, false);
    }
    int col = n * 16 + mrow;
    float bb = b2[col];
    #pragma unroll
    for (int r = 0; r < 8; ++r)
      w_lds[wave][r + hg * 8][col] = acc[r] + bb;
  }
  __syncthreads();

  // ---- SE(3) contraction + scatter: 2 threads per edge --------------------
  const int el   = tid >> 1;            // local edge 0..EDGES_PER_BLOCK-1
  const int part = tid & 1;             // output-j half
  const long long edge = (long long)blockIdx.x * EDGES_PER_BLOCK + el;
  if (edge < n_edges) {
    const float* w = &w_lds[el >> 4][el & 15][0];   // [4][8][8] row-major
    long long src = edge_index[edge];
    long long dst = edge_index[(long long)n_edges + edge];
    const float* nf = node_feats + src * 32;
    float xs0[8], xs1[8][3];
    #pragma unroll
    for (int i = 0; i < 8; ++i) {
      xs0[i] = nf[i];
      xs1[i][0] = nf[8 + i * 3 + 0];
      xs1[i][1] = nf[8 + i * 3 + 1];
      xs1[i][2] = nf[8 + i * 3 + 2];
    }
    const float* sh = edge_sh + edge * 4;
    float sh0 = sh[0], sh1x = sh[1], sh1y = sh[2], sh1z = sh[3];
    float s1[8];
    #pragma unroll
    for (int i = 0; i < 8; ++i)
      s1[i] = xs1[i][0] * sh1x + xs1[i][1] * sh1y + xs1[i][2] * sh1z;

    float* o = out + dst * 32;
    #pragma unroll
    for (int jj = 0; jj < 4; ++jj) {
      int j = part * 4 + jj;
      float a00 = 0.f, a11 = 0.f, a01 = 0.f;
      float a10x = 0.f, a10y = 0.f, a10z = 0.f;
      #pragma unroll
      for (int i = 0; i < 8; ++i) {
        float w00 = w[0 * 64 + i * 8 + j];
        float w01 = w[1 * 64 + i * 8 + j];
        float w10 = w[2 * 64 + i * 8 + j];
        float w11 = w[3 * 64 + i * 8 + j];
        a00 += w00 * xs0[i];
        a11 += w11 * s1[i];
        a01 += w01 * xs0[i];
        a10x += w10 * xs1[i][0];
        a10y += w10 * xs1[i][1];
        a10z += w10 * xs1[i][2];
      }
      atomicAdd(o + j, ALPHA * (a00 * sh0 + INV_SQRT3 * a11));
      atomicAdd(o + 8 + j * 3 + 0, ALPHA * (a01 * sh1x + sh0 * a10x));
      atomicAdd(o + 8 + j * 3 + 1, ALPHA * (a01 * sh1y + sh0 * a10y));
      atomicAdd(o + 8 + j * 3 + 2, ALPHA * (a01 * sh1z + sh0 * a10z));
    }
  }
}

extern "C" void kernel_launch(void* const* d_in, const int* in_sizes, int n_in,
                              void* d_out, int out_size, void* d_ws, size_t ws_size,
                              hipStream_t stream) {
  const float*     node_feats = (const float*)d_in[0];
  const float*     edge_feat  = (const float*)d_in[1];
  const float*     edge_sh    = (const float*)d_in[2];
  const long long* edge_index = (const long long*)d_in[3];
  const float*     mlp_w1     = (const float*)d_in[4];
  const float*     mlp_b1     = (const float*)d_in[5];
  const float*     mlp_w2     = (const float*)d_in[6];
  const float*     mlp_b2     = (const float*)d_in[7];
  const float*     skip_w0    = (const float*)d_in[8];
  const float*     skip_w1    = (const float*)d_in[9];
  float* out = (float*)d_out;

  int n_nodes = in_sizes[0] / 32;
  int n_edges = in_sizes[1] / 16;
  _Float16* wsh = (_Float16*)d_ws;

  se3mp_prep_frags<<<5, 256, 0, stream>>>(mlp_w1, mlp_w2, wsh);
  se3mp_skip<<<(n_nodes + 255) / 256, 256, 0, stream>>>(node_feats, skip_w0,
                                                        skip_w1, out, n_nodes);
  int blocks = (n_edges + EDGES_PER_BLOCK - 1) / EDGES_PER_BLOCK;
  se3mp_edge<<<blocks, EDGES_PER_BLOCK * 2, 0, stream>>>(
      node_feats, edge_feat, edge_sh, edge_index, mlp_b1, mlp_b2, wsh, out,
      n_edges);
}
